// SelfAttentionHead_23648089932252
// MI455X (gfx1250) — matrix operand order repaired and checked by reading, not compiled, not run on hardware
//
#include <hip/hip_runtime.h>
#include <hip/hip_bf16.h>
#include <stdint.h>

// ---------------------------------------------------------------------------
// SelfAttentionHead: B=8, S=2048, D=1024, H=64, fp32 in/out, causal softmax.
// f16 WMMA (v_wmma_f32_16x16x32_f16) everywhere; TDM (tensor_load_to_lds)
// double-buffers K/V tiles into LDS with hardware row padding.
// ---------------------------------------------------------------------------

#define B_  8
#define S_  2048
#define D_  1024
#define H_  64
#define M_  (B_ * S_)    // 16384 rows of x
#define RT  4            // row-tiles (of 16) per projection wave
#define KPITCH 72        // padded LDS row pitch in halfs (64 data + 8 pad)

typedef __attribute__((ext_vector_type(16))) _Float16 v16h;
typedef __attribute__((ext_vector_type(8)))  _Float16 v8h;
typedef __attribute__((ext_vector_type(8)))  float    v8f;

__device__ __forceinline__ v8f wmma_f16f32(v16h a, v16h b, v8f c) {
    // D = A(16x32 f16) * B(32x16 f16) + C(16x16 f32)
    return __builtin_amdgcn_wmma_f32_16x16x32_f16(
        /*neg_a=*/false, a, /*neg_b=*/false, b,
        /*c_mod=*/(short)0, c, /*reuse_a=*/false, /*reuse_b=*/false);
}

// ---------------------------------------------------------------------------
// TDM tile load: DMA one 32x64 f16 tile (row-major) into LDS with 16-byte
// padding after every 128-byte row (=> KPITCH=72 halfs). Descriptor per
// CDNA5 ISA. Guarded so compilation never regresses.
// ---------------------------------------------------------------------------
#if defined(__AMDGCN__) && __has_builtin(__builtin_amdgcn_tensor_load_to_lds)
#define USE_TDM 1
typedef unsigned int u32x4 __attribute__((ext_vector_type(4)));
typedef int          i32x4 __attribute__((ext_vector_type(4)));
typedef int          i32x8 __attribute__((ext_vector_type(8)));

__device__ __forceinline__ void tdm_load_tile_32x64(const _Float16* gsrc,
                                                    unsigned lds_byte_off) {
    unsigned long long ga = (unsigned long long)(size_t)gsrc;
    u32x4 g0;
    g0[0] = 1u;                                   // count=1, user mode, no gather
    g0[1] = lds_byte_off;                         // lds_addr [63:32]
    g0[2] = (unsigned)ga;                         // global_addr [95:64]
    g0[3] = (unsigned)((ga >> 32) & 0x01FFFFFFu)  // global_addr [120:96]
          | (2u << 30);                           // type=2 ("image")
    i32x8 g1;
    // [17:16] data_size=1 (2B); [20] pad_enable; [24:22] pad_interval=4 (32 dw);
    // [31:25] pad_amount=3 (4 dwords = 8 halfs)
    g1[0] = (int)0x07110000;
    g1[1] = (int)(64u << 16);   // tensor_dim0 = 64  (bits 79:48, low half here)
    g1[2] = (int)(32u << 16);   // tensor_dim0 hi = 0; tensor_dim1 = 32 (low16)
    g1[3] = (int)(64u << 16);   // tensor_dim1 hi = 0; tile_dim0 = 64
    g1[4] = 32;                 // tile_dim1 = 32; tile_dim2 = 0
    g1[5] = 64;                 // tensor_dim0_stride = 64 (low 32)
    g1[6] = 0;                  // stride hi / tensor_dim1_stride lo
    g1[7] = 0;
    i32x4 z4 = {0, 0, 0, 0};
#if __clang_major__ >= 23
    i32x8 z8 = {0, 0, 0, 0, 0, 0, 0, 0};
    __builtin_amdgcn_tensor_load_to_lds(g0, g1, z4, z4, z8, 0);
#else
    __builtin_amdgcn_tensor_load_to_lds(g0, g1, z4, z4, 0);
#endif
}
#else
#define USE_TDM 0
#endif

// ---------------------------------------------------------------------------
// Kernel 0: convert fp32 W[D][H] -> f16 Wt[H][D] (transposed: B-frags become
// 16 contiguous halfs).
// ---------------------------------------------------------------------------
__global__ void cvt_w_kernel(const float* __restrict__ W, _Float16* __restrict__ Wt) {
    int i = blockIdx.x * blockDim.x + threadIdx.x;
    if (i < D_ * H_) {
        int d = i >> 6;       // /H_
        int h = i & (H_ - 1); // %H_
        Wt[(size_t)h * D_ + d] = (_Float16)W[i];
    }
}

// ---------------------------------------------------------------------------
// Kernel 1: QKV projection. One wave per 64-row tile; each weight B-fragment
// is reused by RT=4 row tiles (16 WMMAs per k-step, 4x less weight traffic).
// grid = (M_/64, 3); blockIdx.y selects Q/K/V.
// ---------------------------------------------------------------------------
__global__ __launch_bounds__(32) void qkv_proj_kernel(
    const float*    __restrict__ x,
    const _Float16* __restrict__ WtQ,
    const _Float16* __restrict__ WtK,
    const _Float16* __restrict__ WtV,
    _Float16* __restrict__ Q16,
    _Float16* __restrict__ K16,
    _Float16* __restrict__ V16)
{
    const int lane = threadIdx.x & 31;
    const int half = lane >> 4;
    const int lx   = lane & 15;
    const int mb   = blockIdx.x * (16 * RT);

    const _Float16* Wt = (blockIdx.y == 0) ? WtQ : (blockIdx.y == 1) ? WtK : WtV;
    _Float16*       O  = (blockIdx.y == 0) ? Q16 : (blockIdx.y == 1) ? K16 : V16;

    v8f acc[RT][4] = {};

    for (int kk = 0; kk < D_; kk += 32) {
        // A fragments: x rows, K-slice kk..kk+31, fp32 -> f16 inline.
        // lane<16 holds K {0..15}, lane>=16 holds K {16..31}.
        v16h a[RT];
#pragma unroll
        for (int rt = 0; rt < RT; ++rt) {
            const float* xr = x + (size_t)(mb + rt * 16 + lx) * D_ + kk;
            v8f xa = *(const v8f*)(xr + 16 * half);
            v8f xb = *(const v8f*)(xr + 16 * half + 8);
#pragma unroll
            for (int j = 0; j < 8; ++j) {
                a[rt][j]     = (_Float16)xa[j];
                a[rt][8 + j] = (_Float16)xb[j];
            }
        }
#pragma unroll
        for (int nc = 0; nc < 4; ++nc) {
            // B[k][n] = Wt[(nc*16+n)*D + kk + k]; lane: n=lx, k=16*half+j
            const _Float16* wp = Wt + (size_t)(nc * 16 + lx) * D_ + kk + 16 * half;
            v16h bfr = *(const v16h*)wp;
#pragma unroll
            for (int rt = 0; rt < RT; ++rt)
                acc[rt][nc] = wmma_f16f32(a[rt], bfr, acc[rt][nc]);
        }
    }

    // store C tiles as f16 [row][h]; C layout: VGPR i -> row i+8*half
#pragma unroll
    for (int rt = 0; rt < RT; ++rt)
#pragma unroll
        for (int nc = 0; nc < 4; ++nc)
#pragma unroll
            for (int i = 0; i < 8; ++i) {
                int row = rt * 16 + i + 8 * half;
                O[(size_t)(mb + row) * H_ + nc * 16 + lx] = (_Float16)acc[rt][nc][i];
            }
}

// ---------------------------------------------------------------------------
// Kernel 2: causal flash attention. grid = (B, S/64), block = 128 (4 waves).
// Each wave owns 16 queries; K/V tiles stream through double-buffered LDS:
// wave 0 issues the NEXT tile's TDM loads, then waits tensorcnt<=2 so the
// current tile is complete while the next one is still in flight (TENSORcnt
// completes in order; 2 TDM ops per tile).
// ---------------------------------------------------------------------------
__global__ __launch_bounds__(128) void attn_kernel(
    const _Float16* __restrict__ Q16,
    const _Float16* __restrict__ K16,
    const _Float16* __restrict__ V16,
    float* __restrict__ out)
{
    const int b     = blockIdx.x;
    const int qb0   = blockIdx.y * 64;
    const int tid   = threadIdx.x;
    const int wave  = tid >> 5;
    const int lane  = tid & 31;
    const int half  = lane >> 4;
    const int lx    = lane & 15;
    const int qbase = qb0 + wave * 16;

    __shared__ _Float16 sK[2][32 * KPITCH];  // ping-pong [klocal][h], padded
    __shared__ _Float16 sV[2][32 * KPITCH];
    __shared__ _Float16 sP[4][16 * 32];      // per-wave P tile

    const _Float16* kbase_g = K16 + (size_t)b * S_ * H_;
    const _Float16* vbase_g = V16 + (size_t)b * S_ * H_;

    // ----- this wave's Q tile as two A-fragments (H chunks 0-31, 32-63)
    const _Float16* qrow = Q16 + (size_t)(b * S_ + qbase + lx) * H_;
    v16h qa[2];
#pragma unroll
    for (int hc = 0; hc < 2; ++hc) {
        const _Float16* p = qrow + hc * 32 + 16 * half;
        v8h lo = *(const v8h*)p;
        v8h hi = *(const v8h*)(p + 8);
#pragma unroll
        for (int j = 0; j < 8; ++j) { qa[hc][j] = lo[j]; qa[hc][8 + j] = hi[j]; }
    }

    v8f acc[4] = {};
    float rowm[8], rowl[8];
#pragma unroll
    for (int i = 0; i < 8; ++i) { rowm[i] = -__builtin_inff(); rowl[i] = 0.0f; }

    const int   qmax = qbase + 15;
    const int   kend = qb0 + 64;
    const float sc   = 0.125f;               // 1/sqrt(H)

    // ----- preload tile 0 into buffer 0
#if USE_TDM
    if (wave == 0) {
        tdm_load_tile_32x64(kbase_g, (unsigned)(size_t)(&sK[0][0]));
        tdm_load_tile_32x64(vbase_g, (unsigned)(size_t)(&sV[0][0]));
    }
#else
    for (int idx = tid; idx < 32 * H_; idx += 128) {
        int r = idx >> 6, c = idx & (H_ - 1);
        sK[0][r * KPITCH + c] = kbase_g[idx];
        sV[0][r * KPITCH + c] = vbase_g[idx];
    }
#endif

    for (int kb = 0; kb < kend; kb += 32) {
        const int  cur      = (kb >> 5) & 1;
        const bool has_next = (kb + 32) < kend;
#if USE_TDM
        if (wave == 0) {
            if (has_next) {
                tdm_load_tile_32x64(kbase_g + (size_t)(kb + 32) * H_,
                                    (unsigned)(size_t)(&sK[cur ^ 1][0]));
                tdm_load_tile_32x64(vbase_g + (size_t)(kb + 32) * H_,
                                    (unsigned)(size_t)(&sV[cur ^ 1][0]));
                __builtin_amdgcn_s_wait_tensorcnt(2);  // current tile complete
            } else {
                __builtin_amdgcn_s_wait_tensorcnt(0);
            }
        }
#else
        if (has_next) {
            const _Float16* kt = kbase_g + (size_t)(kb + 32) * H_;
            const _Float16* vt = vbase_g + (size_t)(kb + 32) * H_;
            for (int idx = tid; idx < 32 * H_; idx += 128) {
                int r = idx >> 6, c = idx & (H_ - 1);
                sK[cur ^ 1][r * KPITCH + c] = kt[idx];
                sV[cur ^ 1][r * KPITCH + c] = vt[idx];
            }
        }
#endif
        __syncthreads();   // current buffer visible to all waves

        if (kb <= qmax) {
            // ---- scores: c0 = keys kb..kb+15, c1 = keys kb+16..kb+31
            v8f c0 = {}, c1 = {};
#pragma unroll
            for (int hc = 0; hc < 2; ++hc) {
                // B[k][n] = K^T[h][key] = sK[key*KPITCH + h]; k = hc*32+16*half+j
                const _Float16* k0p = &sK[cur][(size_t)lx * KPITCH + hc * 32 + 16 * half];
                const _Float16* k1p = &sK[cur][(size_t)(16 + lx) * KPITCH + hc * 32 + 16 * half];
                v8h b0lo = *(const v8h*)k0p, b0hi = *(const v8h*)(k0p + 8);
                v8h b1lo = *(const v8h*)k1p, b1hi = *(const v8h*)(k1p + 8);
                v16h b0, b1;
#pragma unroll
                for (int j = 0; j < 8; ++j) {
                    b0[j] = b0lo[j]; b0[8 + j] = b0hi[j];
                    b1[j] = b1lo[j]; b1[8 + j] = b1hi[j];
                }
                c0 = wmma_f16f32(qa[hc], b0, c0);
                c1 = wmma_f16f32(qa[hc], b1, c1);
            }
            // ---- online causal softmax (per C row i+8*half)
#pragma unroll
            for (int i = 0; i < 8; ++i) {
                int   row = i + 8 * half;
                int   q   = qbase + row;
                bool  m0  = (kb + lx)      > q;
                bool  m1  = (kb + 16 + lx) > q;
                float s0  = m0 ? -__builtin_inff() : c0[i] * sc;
                float s1  = m1 ? -__builtin_inff() : c1[i] * sc;
                float t   = fmaxf(s0, s1);
#pragma unroll
                for (int off = 1; off < 16; off <<= 1)
                    t = fmaxf(t, __shfl_xor(t, off, 16));
                float mn    = fmaxf(rowm[i], t);
                float scale = __expf(rowm[i] - mn);
                float p0    = m0 ? 0.0f : __expf(s0 - mn);
                float p1    = m1 ? 0.0f : __expf(s1 - mn);
                float rs    = p0 + p1;
#pragma unroll
                for (int off = 1; off < 16; off <<= 1)
                    rs += __shfl_xor(rs, off, 16);
                rowl[i] = rowl[i] * scale + rs;
                rowm[i] = mn;
#pragma unroll
                for (int t4 = 0; t4 < 4; ++t4) acc[t4][i] *= scale;
                sP[wave][row * 32 + lx]      = (_Float16)p0;
                sP[wave][row * 32 + 16 + lx] = (_Float16)p1;
            }
            // ---- reload P as A-fragment (16x32)
            v16h pa;
            {
                const _Float16* pp = &sP[wave][lx * 32 + 16 * half];
                v8h lo = *(const v8h*)pp;
                v8h hi = *(const v8h*)(pp + 8);
#pragma unroll
                for (int j = 0; j < 8; ++j) { pa[j] = lo[j]; pa[8 + j] = hi[j]; }
            }
            // ---- acc += P(16x32) * V(32x64): B[k][n] = sV[k*KPITCH + nc*16+n]
#pragma unroll
            for (int nc = 0; nc < 4; ++nc) {
                v16h bv;
#pragma unroll
                for (int j = 0; j < 16; ++j)
                    bv[j] = sV[cur][(size_t)(16 * half + j) * KPITCH + nc * 16 + lx];
                acc[nc] = wmma_f16f32(pa, bv, acc[nc]);
            }
        }
        __syncthreads();   // everyone done with 'cur' before it is refilled
    }

    // ---- epilogue: out[b][q][h] = acc / l
#pragma unroll
    for (int i = 0; i < 8; ++i) {
        int   row = i + 8 * half;
        int   q   = qbase + row;
        float inv = 1.0f / rowl[i];
#pragma unroll
        for (int nc = 0; nc < 4; ++nc)
            out[(size_t)(b * S_ + q) * H_ + nc * 16 + lx] = acc[nc][i] * inv;
    }
}

// ---------------------------------------------------------------------------
// Host-side launcher
// ---------------------------------------------------------------------------
extern "C" void kernel_launch(void* const* d_in, const int* in_sizes, int n_in,
                              void* d_out, int out_size, void* d_ws, size_t ws_size,
                              hipStream_t stream) {
    const float* x  = (const float*)d_in[0];
    const float* Wq = (const float*)d_in[1];
    const float* Wk = (const float*)d_in[2];
    const float* Wv = (const float*)d_in[3];
    float* out = (float*)d_out;

    // Workspace layout (f16): 3 transposed weights + Q/K/V activations (~6.4 MB)
    char* ws = (char*)d_ws;
    _Float16* WtQ = (_Float16*)ws;
    _Float16* WtK = WtQ + (size_t)H_ * D_;
    _Float16* WtV = WtK + (size_t)H_ * D_;
    _Float16* Q16 = WtV + (size_t)H_ * D_;
    _Float16* K16 = Q16 + (size_t)M_ * H_;
    _Float16* V16 = K16 + (size_t)M_ * H_;

    {
        dim3 g((D_ * H_ + 255) / 256), t(256);
        cvt_w_kernel<<<g, t, 0, stream>>>(Wq, WtQ);
        cvt_w_kernel<<<g, t, 0, stream>>>(Wk, WtK);
        cvt_w_kernel<<<g, t, 0, stream>>>(Wv, WtV);
    }
    qkv_proj_kernel<<<dim3(M_ / (16 * RT), 3), 32, 0, stream>>>(
        x, WtQ, WtK, WtV, Q16, K16, V16);
    attn_kernel<<<dim3(B_, S_ / 64), 128, 0, stream>>>(Q16, K16, V16, out);
}
